// RichGCNModel_28132035788997
// MI455X (gfx1250) — matrix operand order repaired and checked by reading, not compiled, not run on hardware
//
#include <hip/hip_runtime.h>
#include <hip/hip_bf16.h>

typedef _Float16 v16h __attribute__((ext_vector_type(16)));
typedef float    v8f  __attribute__((ext_vector_type(8)));

#define NN 100000
#define EE 3200000
#define GG 2000

// ---------------- degree / normalization ----------------
__global__ void deg_init_kernel(float* deg, int n) {
    int i = blockIdx.x * blockDim.x + threadIdx.x;
    if (i < n) deg[i] = 1.0f;                 // self loop
}
__global__ void deg_count_kernel(const int* __restrict__ dst, float* deg, int e) {
    int i = blockIdx.x * blockDim.x + threadIdx.x;
    if (i < e) atomicAdd(&deg[dst[i]], 1.0f);
}
__global__ void dinv_kernel(float* deg, int n) {
    int i = blockIdx.x * blockDim.x + threadIdx.x;
    if (i < n) deg[i] = rsqrtf(fmaxf(deg[i], 1.0f));   // in-place deg -> deg^-1/2
}

// ---------------- WMMA GEMM, layer 1: [N,10] @ [10,64], epilogue scale by dinv ----------------
// block = 128 threads (4 waves), wave w -> output cols [16w, 16w+16)
__global__ __launch_bounds__(128) void gemm1_wmma(const float* __restrict__ x,
                                                  const float* __restrict__ W,
                                                  const float* __restrict__ dinv,
                                                  float* __restrict__ outA,
                                                  float* __restrict__ outB) {
    const int Kin = 10, Fout = 64;
    int lane = threadIdx.x & 31;
    int wave = threadIdx.x >> 5;
    int mtile = blockIdx.x;
    int row  = lane & 15;
    int half = lane >> 4;
    int m = mtile * 16 + row;

    // A: 16x32 f16; lane holds row m; elems 0..7 -> K=half*8+e, elems 8..15 -> K=16+half*8+e (all >= Kin here)
    v16h a;
#pragma unroll
    for (int e = 0; e < 8; e++) {
        int k = half * 8 + e;
        a[e] = (k < Kin) ? (_Float16)x[m * Kin + k] : (_Float16)0.f;
    }
#pragma unroll
    for (int e = 0; e < 8; e++) a[8 + e] = (_Float16)0.f;

    // B: 32x16 f16; lane holds col; elems 0..7 -> K=half*8+e, elems 8..15 -> K=16+half*8+e
    int col = wave * 16 + row;
    int kb  = half * 8;
    v16h b;
#pragma unroll
    for (int e = 0; e < 16; e++) {
        int k = kb + e + (e < 8 ? 0 : 8);
        b[e] = (k < Kin) ? (_Float16)W[k * Fout + col] : (_Float16)0.f;
    }

    v8f c = {};
    c = __builtin_amdgcn_wmma_f32_16x16x32_f16(false, a, false, b, (short)0, c, false, false);

#pragma unroll
    for (int r = 0; r < 8; r++) {
        int mm = mtile * 16 + half * 8 + r;
        float v = c[r] * dinv[mm];            // pre-scale row by deg^-1/2
        outA[mm * Fout + col] = v;            // hs (gather source)
        outB[mm * Fout + col] = v;            // acc init = self-loop term
    }
}

// ---------------- WMMA GEMM, layer 2: [N,64] @ [64,128], K = 2 WMMA steps ----------------
// block = 256 threads (8 waves), wave w -> output cols [16w, 16w+16)
__global__ __launch_bounds__(256) void gemm2_wmma(const float* __restrict__ x,
                                                  const float* __restrict__ W,
                                                  const float* __restrict__ dinv,
                                                  float* __restrict__ outA,
                                                  float* __restrict__ outB) {
    const int Kin = 64, Fout = 128;
    int lane = threadIdx.x & 31;
    int wave = threadIdx.x >> 5;
    int mtile = blockIdx.x;
    int row  = lane & 15;
    int half = lane >> 4;
    int m   = mtile * 16 + row;
    int col = wave * 16 + row;

    v8f c = {};
#pragma unroll
    for (int ks = 0; ks < 2; ks++) {
        int kbA = ks * 32;
        v16h a;
#pragma unroll
        for (int e = 0; e < 8; e++) a[e]     = (_Float16)x[m * Kin + kbA + half * 8 + e];
#pragma unroll
        for (int e = 0; e < 8; e++) a[8 + e] = (_Float16)x[m * Kin + kbA + 16 + half * 8 + e];
        v16h b;
        int kbB = kbA + half * 8;
#pragma unroll
        for (int e = 0; e < 16; e++) b[e] = (_Float16)W[(kbB + e + (e < 8 ? 0 : 8)) * Fout + col];
        c = __builtin_amdgcn_wmma_f32_16x16x32_f16(false, a, false, b, (short)0, c, false, false);
    }

#pragma unroll
    for (int r = 0; r < 8; r++) {
        int mm = mtile * 16 + half * 8 + r;
        float v = c[r] * dinv[mm];
        outA[mm * Fout + col] = v;
        outB[mm * Fout + col] = v;
    }
}

// ---------------- edge scatter-add: acc[dst] += hs[src]  (float4 per thread) ----------------
__global__ void edge_agg_kernel(const int* __restrict__ src, const int* __restrict__ dst,
                                const float* __restrict__ hs, float* __restrict__ acc,
                                int e, int F) {
    long long t = (long long)blockIdx.x * blockDim.x + threadIdx.x;
    int chunks = F >> 2;
    long long total = (long long)e * chunks;
    if (t >= total) return;
    int ei = (int)(t / chunks);
    int c  = (int)(t % chunks) << 2;
    int s = src[ei], d = dst[ei];
    const float4 v = *(const float4*)(hs + (size_t)s * F + c);
    float* p = acc + (size_t)d * F + c;
    atomicAdd(p + 0, v.x);
    atomicAdd(p + 1, v.y);
    atomicAdd(p + 2, v.z);
    atomicAdd(p + 3, v.w);
}

// ---------------- post-aggregation: out = dinv[i]*acc + bias[f] ----------------
__global__ void finish_kernel(const float* __restrict__ acc, const float* __restrict__ dinv,
                              const float* __restrict__ bias, float* __restrict__ out,
                              int n, int F) {
    int idx = blockIdx.x * blockDim.x + threadIdx.x;
    if (idx >= n * F) return;
    int i = idx / F;
    int f = idx - i * F;
    out[idx] = acc[idx] * dinv[i] + bias[f];
}

// ---------------- BN stats: per-feature sum & sumsq (block partial reduce + atomics) ----------------
__global__ void stats_kernel(const float* __restrict__ h, float* __restrict__ sums,
                             int n, int F) {
    int t = threadIdx.x;                 // blockDim = 256
    int f = t & (F - 1);
    int rowsPer = 256 / F;
    int sub = t / F;
    int chunk = (n + gridDim.x - 1) / gridDim.x;
    int i0 = blockIdx.x * chunk;
    int i1 = min(i0 + chunk, n);
    float s = 0.f, q = 0.f;
    for (int i = i0 + sub; i < i1; i += rowsPer) {
        float v = h[(size_t)i * F + f];
        s += v; q += v * v;
    }
    __shared__ float ls[256], lq[256];
    ls[t] = s; lq[t] = q;
    __syncthreads();
    if (sub == 0) {
        for (int r = 1; r < rowsPer; r++) { s += ls[t + r * F]; q += lq[t + r * F]; }
        atomicAdd(&sums[f], s);
        atomicAdd(&sums[F + f], q);
    }
}

// ---------------- BN apply + ReLU ----------------
__global__ void bnrelu_kernel(const float* __restrict__ hin, const float* __restrict__ sums,
                              const float* __restrict__ gamma, const float* __restrict__ beta,
                              float* __restrict__ hout, int n, int F) {
    int idx = blockIdx.x * blockDim.x + threadIdx.x;
    if (idx >= n * F) return;
    int f = idx % F;
    float inv_n = 1.0f / (float)n;
    float mu  = sums[f] * inv_n;
    float var = sums[F + f] * inv_n - mu * mu;
    float v = (hin[idx] - mu) * rsqrtf(var + 1e-5f) * gamma[f] + beta[f];
    hout[idx] = fmaxf(v, 0.f);
}

// ---------------- pooling ----------------
__global__ void zero_kernel(float* p, int n) {
    int i = blockIdx.x * blockDim.x + threadIdx.x;
    if (i < n) p[i] = 0.f;
}
__global__ void count_kernel(const int* __restrict__ batch, float* cnt, int n) {
    int i = blockIdx.x * blockDim.x + threadIdx.x;
    if (i < n) atomicAdd(&cnt[batch[i]], 1.0f);
}
__global__ void pool_kernel(const int* __restrict__ batch, const float* __restrict__ h,
                            float* __restrict__ pooled, int n) {
    int idx = blockIdx.x * blockDim.x + threadIdx.x;
    if (idx >= n * 128) return;
    int i = idx >> 7;
    int f = idx & 127;
    atomicAdd(&pooled[(size_t)batch[i] * 128 + f], h[idx]);
}

// ---------------- head MLP: [G,128] -> relu 64 -> 1 ----------------
__global__ __launch_bounds__(64) void head_kernel(const float* __restrict__ pooled,
                                                  const float* __restrict__ cnt,
                                                  const float* __restrict__ fW1,
                                                  const float* __restrict__ fb1,
                                                  const float* __restrict__ fW2,
                                                  const float* __restrict__ fb2,
                                                  float* __restrict__ out) {
    int g = blockIdx.x;
    int t = threadIdx.x;                 // 64 hidden units
    float inv = 1.0f / fmaxf(cnt[g], 1.0f);
    float acc = 0.f;
    for (int k = 0; k < 128; k++)
        acc += pooled[(size_t)g * 128 + k] * inv * fW1[k * 64 + t];
    float hid = fmaxf(acc + fb1[t], 0.f);
    __shared__ float red[64];
    red[t] = hid * fW2[t];
    __syncthreads();
    for (int s = 32; s > 0; s >>= 1) {
        if (t < s) red[t] += red[t + s];
        __syncthreads();
    }
    if (t == 0) out[g] = red[0] + fb2[0];
}

static inline unsigned gridFor(long long total, int block) {
    return (unsigned)((total + block - 1) / block);
}

extern "C" void kernel_launch(void* const* d_in, const int* in_sizes, int n_in,
                              void* d_out, int out_size, void* d_ws, size_t ws_size,
                              hipStream_t stream) {
    (void)in_sizes; (void)n_in; (void)out_size; (void)ws_size;
    const float* x     = (const float*)d_in[0];
    const int*   src   = (const int*)d_in[1];
    const int*   dst   = (const int*)d_in[2];
    const int*   batch = (const int*)d_in[3];
    const float* W1  = (const float*)d_in[4];
    const float* b1  = (const float*)d_in[5];
    const float* g1  = (const float*)d_in[6];
    const float* be1 = (const float*)d_in[7];
    const float* W2  = (const float*)d_in[8];
    const float* b2  = (const float*)d_in[9];
    const float* g2  = (const float*)d_in[10];
    const float* be2 = (const float*)d_in[11];
    const float* fW1 = (const float*)d_in[12];
    const float* fb1 = (const float*)d_in[13];
    const float* fW2 = (const float*)d_in[14];
    const float* fb2 = (const float*)d_in[15];
    float* out = (float*)d_out;

    // workspace layout (floats)
    float* ws = (float*)d_ws;
    size_t off = 0;
    float* dinv   = ws + off; off += NN;                 // deg -> deg^-1/2 in place
    float* bufA   = ws + off; off += (size_t)NN * 64;    // hs1 / h1 post-agg
    float* bufB   = ws + off; off += (size_t)NN * 64;    // acc1 / x2
    float* bufC   = ws + off; off += (size_t)NN * 128;   // hs2 / h2 post-agg
    float* bufD   = ws + off; off += (size_t)NN * 128;   // acc2 / h2 final
    float* stats  = ws + off; off += 256;                // sum|sumsq, up to F=128
    float* pooled = ws + off; off += (size_t)GG * 128;
    float* cnt    = ws + off; off += GG;                 // contiguous after pooled

    // ---- degrees & symmetric norm ----
    deg_init_kernel<<<gridFor(NN, 256), 256, 0, stream>>>(dinv, NN);
    deg_count_kernel<<<gridFor(EE, 256), 256, 0, stream>>>(dst, dinv, EE);
    dinv_kernel<<<gridFor(NN, 256), 256, 0, stream>>>(dinv, NN);

    // ---- layer 1: GCNConv(10->64) + BN + ReLU ----
    gemm1_wmma<<<NN / 16, 128, 0, stream>>>(x, W1, dinv, bufA, bufB);
    edge_agg_kernel<<<gridFor((long long)EE * 16, 256), 256, 0, stream>>>(src, dst, bufA, bufB, EE, 64);
    finish_kernel<<<gridFor((long long)NN * 64, 256), 256, 0, stream>>>(bufB, dinv, b1, bufA, NN, 64);
    zero_kernel<<<1, 256, 0, stream>>>(stats, 256);
    stats_kernel<<<256, 256, 0, stream>>>(bufA, stats, NN, 64);
    bnrelu_kernel<<<gridFor((long long)NN * 64, 256), 256, 0, stream>>>(bufA, stats, g1, be1, bufB, NN, 64);

    // ---- layer 2: GCNConv(64->128) + BN + ReLU ----
    gemm2_wmma<<<NN / 16, 256, 0, stream>>>(bufB, W2, dinv, bufC, bufD);
    edge_agg_kernel<<<gridFor((long long)EE * 32, 256), 256, 0, stream>>>(src, dst, bufC, bufD, EE, 128);
    finish_kernel<<<gridFor((long long)NN * 128, 256), 256, 0, stream>>>(bufD, dinv, b2, bufC, NN, 128);
    zero_kernel<<<1, 256, 0, stream>>>(stats, 256);
    stats_kernel<<<256, 256, 0, stream>>>(bufC, stats, NN, 128);
    bnrelu_kernel<<<gridFor((long long)NN * 128, 256), 256, 0, stream>>>(bufC, stats, g2, be2, bufD, NN, 128);

    // ---- global mean pool + head MLP ----
    zero_kernel<<<gridFor(GG * 128 + GG, 256), 256, 0, stream>>>(pooled, GG * 128 + GG);
    count_kernel<<<gridFor(NN, 256), 256, 0, stream>>>(batch, cnt, NN);
    pool_kernel<<<gridFor((long long)NN * 128, 256), 256, 0, stream>>>(batch, bufD, pooled, NN);
    head_kernel<<<GG, 64, 0, stream>>>(pooled, cnt, fW1, fb1, fW2, fb2, out);
}
